// MambaNet_66425964200019
// MI455X (gfx1250) — compile-verified, matrix operand order checked by reading, not run on hardware
//
#include <hip/hip_runtime.h>

// ---------------------------------------------------------------------------
// MambaNet forward for MI455X (gfx1250, wave32, WMMA).
// Dense projections on v_wmma_f32_16x16x32_f16; GEMM tiles staged through LDS
// (async global->LDS when the toolchain exposes the gfx1250 builtins).
// ---------------------------------------------------------------------------

typedef _Float16 v16h __attribute__((ext_vector_type(16)));
typedef _Float16 h8_t __attribute__((ext_vector_type(8)));
typedef float    v8f  __attribute__((ext_vector_type(8)));

constexpr int Bc   = 4;
constexpr int Sc   = 512;
constexpr int DMc  = 768;
constexpr int Lc   = 2;
constexpr int DIc  = 2 * DMc;          // 1536
constexpr int DSc  = 16;
constexpr int Kc   = 4;
constexpr int DTRc = 48;
constexpr int DTRp = 64;               // padded K for dt GEMM
constexpr int DBCc = DTRc + 2 * DSc;   // 80
constexpr int BSc  = Bc * Sc;          // 2048
constexpr int FFH  = 2 * DMc;          // 1536

#ifndef __has_builtin
#define __has_builtin(x) 0
#endif
#if __has_builtin(__builtin_amdgcn_global_load_async_to_lds_b128) && \
    __has_builtin(__builtin_amdgcn_s_wait_asynccnt)
#define USE_ASYNC_LDS 1
#else
#define USE_ASYNC_LDS 0
#endif

#define DEV __device__ __forceinline__

DEV float sigmf(float x)     { return 1.0f / (1.0f + __expf(-x)); }
DEV float siluf(float x)     { return x * sigmf(x); }
DEV float geluf(float x)     { return 0.5f * x * (1.0f + erff(x * 0.70710678118654752f)); }
DEV float softplusf(float x) { return (x > 20.0f) ? x : log1pf(__expf(x)); }

DEV v16h cat16(h8_t lo, h8_t hi) {
  return __builtin_shufflevector(lo, hi, 0, 1, 2, 3, 4, 5, 6, 7,
                                 8, 9, 10, 11, 12, 13, 14, 15);
}

#if USE_ASYNC_LDS
// Builtin signature (from hipcc diagnostic): param 0 is
// 'int __vector(4) __device__ *' (addrspace(1)); param 1 is the LDS side.
typedef int v4i_vs __attribute__((vector_size(16)));
typedef __attribute__((address_space(1))) v4i_vs* gv4i_p;
typedef __attribute__((address_space(3))) v4i_vs* lv4i_p;

DEV void async_copy16(const _Float16* g, _Float16* l) {
  __builtin_amdgcn_global_load_async_to_lds_b128((gv4i_p)g, (lv4i_p)l, 0, 0);
}
#endif

// ---------------------------------------------------------------------------
// LDS-tiled WMMA GEMM: out[m,n] = epi( sum_k A16[m,k]*W16[n,k] + bias[n] )
// Requires: M % 64 == 0, N % 128 == 0, Kd % 32 == 0.
// Block = 256 thr = 8 waves; block tile 64(M) x 128(N); K stepped by 32.
// A tile 64x32 f16 (4KB) + B tile 128x32 f16 (8KB) staged per step.
// EPI: 0 none, 1 +bias, 2 +bias,gelu, 3 +bias,softplus
// ---------------------------------------------------------------------------
template <int EPI>
__global__ __launch_bounds__(256) void wmma_gemm_lds(
    const _Float16* __restrict__ A, int lda,
    const _Float16* __restrict__ W, int ldw,
    float* __restrict__ out32, _Float16* __restrict__ out16,
    const float* __restrict__ bias, int N, int Kd)
{
  __shared__ _Float16 As[64 * 32];
  __shared__ _Float16 Bs[128 * 32];

  const int t    = threadIdx.x;
  const int lane = t & 31;
  const int wid  = t >> 5;
  const int g    = lane >> 4;   // 0: lanes 0-15, 1: lanes 16-31
  const int hl   = lane & 15;
  const int waveM = (wid & 3) * 16;
  const int waveN = (wid >> 2) * 64;
  const int blockM = blockIdx.y * 64;
  const int blockN = blockIdx.x * 128;

  // Tile loader mapping: thread -> one 16B chunk of A, two 16B chunks of B.
  const int lrow = t >> 2;            // 0..63
  const int lcol = (t & 3) * 8;       // 0,8,16,24
  const _Float16* aSrc  = A + (size_t)(blockM + lrow) * lda + lcol;
  const _Float16* bSrc0 = W + (size_t)(blockN + lrow) * ldw + lcol;
  const _Float16* bSrc1 = W + (size_t)(blockN + 64 + lrow) * ldw + lcol;
  _Float16* aDst  = &As[lrow * 32 + lcol];
  _Float16* bDst0 = &Bs[lrow * 32 + lcol];
  _Float16* bDst1 = &Bs[(64 + lrow) * 32 + lcol];

  const _Float16* aFragP = &As[(waveM + hl) * 32 + 8 * g];

  v8f acc[4] = {};

  for (int k0 = 0; k0 < Kd; k0 += 32) {
#if USE_ASYNC_LDS
    async_copy16(aSrc + k0, aDst);
    async_copy16(bSrc0 + k0, bDst0);
    async_copy16(bSrc1 + k0, bDst1);
    __builtin_amdgcn_s_wait_asynccnt(0);
#else
    *(h8_t*)aDst  = *(const h8_t*)(aSrc + k0);
    *(h8_t*)bDst0 = *(const h8_t*)(bSrc0 + k0);
    *(h8_t*)bDst1 = *(const h8_t*)(bSrc1 + k0);
#endif
    __syncthreads();

    // A fragment (16x32 f16, ISA 7.12.2): lane row = hl;
    // halves 0..7 -> K = 8*g + j ; halves 8..15 -> K = 16 + 8*g + j.
    const v16h af = cat16(*(const h8_t*)aFragP, *(const h8_t*)(aFragP + 16));

#pragma unroll
    for (int sub = 0; sub < 4; ++sub) {
      const _Float16* bp = &Bs[(waveN + sub * 16 + hl) * 32 + 8 * g];
      const v16h bf = cat16(*(const h8_t*)bp, *(const h8_t*)(bp + 16));
      acc[sub] = __builtin_amdgcn_wmma_f32_16x16x32_f16(
          false, af, false, bf, (short)0, acc[sub], false, false);
    }
    __syncthreads();
  }

  // C/D layout: lane l, vgpr v -> row = 8*(l>>4)+v, col = l&15
#pragma unroll
  for (int sub = 0; sub < 4; ++sub) {
    const int n = blockN + waveN + sub * 16 + hl;
    float bv = 0.0f;
    if (EPI >= 1) bv = bias[n];
#pragma unroll
    for (int v = 0; v < 8; ++v) {
      const size_t m = (size_t)(blockM + waveM + 8 * g + v);
      float r = acc[sub][v];
      if (EPI >= 1) r += bv;
      if (EPI == 2) r = geluf(r);
      if (EPI == 3) r = softplusf(r);
      if (out32) out32[m * (size_t)N + n] = r;
      if (out16) out16[m * (size_t)N + n] = (_Float16)r;
    }
  }
}

// ---------------------------------------------------------------------------
// Direct-global WMMA GEMM with N-edge guarding (used only for N=80 x-proj).
// ---------------------------------------------------------------------------
template <int EPI>
__global__ __launch_bounds__(256) void wmma_gemm_edge(
    const _Float16* __restrict__ A, int lda,
    const _Float16* __restrict__ W, int ldw,
    float* __restrict__ out32, _Float16* __restrict__ out16,
    const float* __restrict__ bias, int N, int Kd)
{
  const int lane = threadIdx.x & 31;
  const int wid  = threadIdx.x >> 5;
  const int g    = lane >> 4;
  const int hl   = lane & 15;
  const int tileM  = blockIdx.y * 64 + (wid & 3) * 16;
  const int tileN0 = blockIdx.x * 128 + (wid >> 2) * 64;

  v8f acc[4] = {};
  const _Float16* aRow = A + (size_t)(tileM + hl) * lda;

  for (int k0 = 0; k0 < Kd; k0 += 32) {
    __builtin_prefetch(aRow + k0 + 32, 0, 1);  // speculative: OOB dropped by HW
    const v16h af = cat16(*(const h8_t*)(aRow + k0 + 8 * g),
                          *(const h8_t*)(aRow + k0 + 16 + 8 * g));
#pragma unroll
    for (int sub = 0; sub < 4; ++sub) {
      const int n = tileN0 + sub * 16 + hl;
      v16h bf;
      if (n < N) {
        const _Float16* wRow = W + (size_t)n * ldw;
        bf = cat16(*(const h8_t*)(wRow + k0 + 8 * g),
                   *(const h8_t*)(wRow + k0 + 16 + 8 * g));
      } else {
        bf = v16h{};
      }
      acc[sub] = __builtin_amdgcn_wmma_f32_16x16x32_f16(
          false, af, false, bf, (short)0, acc[sub], false, false);
    }
  }

#pragma unroll
  for (int sub = 0; sub < 4; ++sub) {
    const int n = tileN0 + sub * 16 + hl;
    if (n >= N) continue;
    float bv = 0.0f;
    if (EPI >= 1) bv = bias[n];
#pragma unroll
    for (int v = 0; v < 8; ++v) {
      const size_t m = (size_t)(tileM + 8 * g + v);
      float r = acc[sub][v];
      if (EPI >= 1) r += bv;
      if (EPI == 2) r = geluf(r);
      if (EPI == 3) r = softplusf(r);
      if (out32) out32[m * (size_t)N + n] = r;
      if (out16) out16[m * (size_t)N + n] = (_Float16)r;
    }
  }
}

// ---------------------------------------------------------------------------
// LayerNorm over D per row; 256 threads/row; wave32 shfl + shared reduction.
// ---------------------------------------------------------------------------
template <bool OUT16>
__global__ __launch_bounds__(256) void layernorm_k(
    const float* __restrict__ x, const float* __restrict__ w,
    const float* __restrict__ b, _Float16* __restrict__ o16,
    float* __restrict__ o32, int D)
{
  const int row = blockIdx.x;
  const float* xr = x + (size_t)row * D;
  float s = 0.0f, s2 = 0.0f;
  for (int i = threadIdx.x; i < D; i += 256) { float v = xr[i]; s += v; s2 += v * v; }
#pragma unroll
  for (int o = 16; o > 0; o >>= 1) { s += __shfl_down(s, o, 32); s2 += __shfl_down(s2, o, 32); }
  __shared__ float sh[18];
  const int lane = threadIdx.x & 31, wid = threadIdx.x >> 5;
  if (lane == 0) { sh[wid] = s; sh[8 + wid] = s2; }
  __syncthreads();
  if (threadIdx.x == 0) {
    float ts = 0.0f, ts2 = 0.0f;
    for (int i = 0; i < 8; ++i) { ts += sh[i]; ts2 += sh[8 + i]; }
    const float mu = ts / (float)D;
    sh[16] = mu;
    sh[17] = rsqrtf(ts2 / (float)D - mu * mu + 1e-5f);
  }
  __syncthreads();
  const float mu = sh[16], rs = sh[17];
  for (int i = threadIdx.x; i < D; i += 256) {
    const float v = (xr[i] - mu) * rs * w[i] + b[i];
    if (OUT16) o16[(size_t)row * D + i] = (_Float16)v;
    else       o32[(size_t)row * D + i] = v;
  }
}

// ---------------------------------------------------------------------------
// Small elementwise / staging kernels
// ---------------------------------------------------------------------------
__global__ void f32_to_f16_k(const float* __restrict__ s, _Float16* __restrict__ d, int n) {
  for (int i = blockIdx.x * blockDim.x + threadIdx.x; i < n; i += gridDim.x * blockDim.x)
    d[i] = (_Float16)s[i];
}

__global__ void wdt_pad_k(const float* __restrict__ s, _Float16* __restrict__ d) {
  int i = blockIdx.x * blockDim.x + threadIdx.x;
  if (i >= DIc * DTRp) return;
  const int r = i / DTRp, c = i % DTRp;
  d[i] = (c < DTRc) ? (_Float16)s[r * DTRc + c] : (_Float16)0.0f;
}

__global__ void dt_pad_k(const float* __restrict__ dbc, _Float16* __restrict__ d) {
  int i = blockIdx.x * blockDim.x + threadIdx.x;
  if (i >= BSc * DTRp) return;
  const int r = i / DTRp, c = i % DTRp;
  d[i] = (c < DTRc) ? (_Float16)dbc[(size_t)r * DBCc + c] : (_Float16)0.0f;
}

__global__ void embed_k(const float* __restrict__ x, const float* __restrict__ w_in,
                        const float* __restrict__ b_in, const float* __restrict__ pe,
                        float* __restrict__ h) {
  int i = blockIdx.x * blockDim.x + threadIdx.x;
  if (i >= BSc * DMc) return;
  const int row = i / DMc, d = i % DMc, s = row % Sc;
  h[i] = x[row] * w_in[d] + b_in[d] + pe[(size_t)s * DMc + d];
}

__global__ void conv_silu_k(const float* __restrict__ xz, const float* __restrict__ cw,
                            const float* __restrict__ cb, float* __restrict__ xc) {
  int i = blockIdx.x * blockDim.x + threadIdx.x;
  if (i >= BSc * DIc) return;
  const int row = i / DIc, d = i % DIc;
  const int b = row / Sc, s = row % Sc;
  float acc = cb[d];
#pragma unroll
  for (int j = 0; j < Kc; ++j) {
    const int ss = s - (Kc - 1) + j;
    if (ss >= 0) acc += xz[((size_t)(b * Sc + ss)) * (2 * DIc) + d] * cw[d * Kc + j];
  }
  xc[i] = siluf(acc);
}

__global__ void add_inplace_k(float* __restrict__ dst, const float* __restrict__ src, int n) {
  int i = blockIdx.x * blockDim.x + threadIdx.x;
  if (i < n) dst[i] += src[i];
}

// ---------------------------------------------------------------------------
// Selective scan: one thread per (b, d) channel carries 16 states over S=512.
// Fuses +u*D, SiLU(z) gating, and writes y directly as f16 (out-proj A input).
// ---------------------------------------------------------------------------
__global__ __launch_bounds__(256) void scan_k(
    const float* __restrict__ delta, const float* __restrict__ dbc,
    const float* __restrict__ xz, const float* __restrict__ xc,
    const float* __restrict__ A_log, const float* __restrict__ Dp,
    _Float16* __restrict__ y16)
{
  int idx = blockIdx.x * blockDim.x + threadIdx.x;
  if (idx >= Bc * DIc) return;
  const int b = idx / DIc, d = idx % DIc;
  float Areg[DSc];
#pragma unroll
  for (int j = 0; j < DSc; ++j) Areg[j] = -__expf(A_log[d * DSc + j]);
  const float Dd = Dp[d];
  float hst[DSc];
#pragma unroll
  for (int j = 0; j < DSc; ++j) hst[j] = 0.0f;

  for (int s = 0; s < Sc; ++s) {
    const size_t r = (size_t)(b * Sc + s);
    const float dlt = delta[r * DIc + d];
    const float u   = xc[r * DIc + d];
    const float du  = dlt * u;
    const float* bc = dbc + r * DBCc;
    float y = 0.0f;
#pragma unroll
    for (int j = 0; j < DSc; ++j) {
      hst[j] = hst[j] * __expf(dlt * Areg[j]) + du * bc[DTRc + j];
      y += hst[j] * bc[DTRc + DSc + j];
    }
    y += u * Dd;
    const float zv = xz[r * (2 * DIc) + DIc + d];
    y *= siluf(zv);
    y16[r * DIc + d] = (_Float16)y;
  }
}

__global__ __launch_bounds__(256) void pool_head_k(
    const float* __restrict__ hf, const float* __restrict__ wh,
    const float* __restrict__ bh, float* __restrict__ out)
{
  const int b = blockIdx.x;
  float acc = 0.0f;
  for (int i = threadIdx.x; i < Sc * DMc; i += 256)
    acc += hf[(size_t)b * Sc * DMc + i] * wh[i % DMc];
#pragma unroll
  for (int o = 16; o > 0; o >>= 1) acc += __shfl_down(acc, o, 32);
  __shared__ float sh[8];
  if ((threadIdx.x & 31) == 0) sh[threadIdx.x >> 5] = acc;
  __syncthreads();
  if (threadIdx.x == 0) {
    float t = 0.0f;
    for (int i = 0; i < 8; ++i) t += sh[i];
    out[b] = t / (float)Sc + bh[0];
  }
}

// ---------------------------------------------------------------------------
// Host driver
// ---------------------------------------------------------------------------
extern "C" void kernel_launch(void* const* d_in, const int* in_sizes, int n_in,
                              void* d_out, int out_size, void* d_ws, size_t ws_size,
                              hipStream_t stream) {
  (void)in_sizes; (void)n_in; (void)out_size; (void)ws_size;
  const float* x        = (const float*)d_in[0];
  const float* w_in     = (const float*)d_in[1];
  const float* b_in     = (const float*)d_in[2];
  const float* pe       = (const float*)d_in[3];
  const float* ln1_w    = (const float*)d_in[4];
  const float* ln1_b    = (const float*)d_in[5];
  const float* w_inproj = (const float*)d_in[6];
  const float* conv_w   = (const float*)d_in[7];
  const float* conv_b   = (const float*)d_in[8];
  const float* w_xproj  = (const float*)d_in[9];
  const float* w_dt     = (const float*)d_in[10];
  const float* b_dt     = (const float*)d_in[11];
  const float* A_log    = (const float*)d_in[12];
  const float* Dp       = (const float*)d_in[13];
  const float* w_outproj= (const float*)d_in[14];
  const float* ln2_w    = (const float*)d_in[15];
  const float* ln2_b    = (const float*)d_in[16];
  const float* ff_w1    = (const float*)d_in[17];
  const float* ff_b1    = (const float*)d_in[18];
  const float* ff_w2    = (const float*)d_in[19];
  const float* ff_b2    = (const float*)d_in[20];
  const float* lnf_w    = (const float*)d_in[21];
  const float* lnf_b    = (const float*)d_in[22];
  const float* w_head   = (const float*)d_in[23];
  const float* b_head   = (const float*)d_in[24];

  // ---- workspace layout ----
  constexpr size_t PLH = (size_t)2 * DIc * DMc + (size_t)DBCc * DIc + (size_t)DIc * DTRp
                       + (size_t)DMc * DIc + (size_t)FFH * DMc + (size_t)DMc * FFH;
  char* ws = (char*)d_ws;
  size_t off = 0;
  auto take = [&](size_t bytes) -> char* {
    char* p = ws + off; off = (off + bytes + 255) & ~(size_t)255; return p;
  };
  _Float16* wf16 = (_Float16*)take(Lc * PLH * sizeof(_Float16));
  float*    hbuf = (float*)   take((size_t)BSc * DMc * 4);
  _Float16* a16  = (_Float16*)take((size_t)BSc * DIc * 2);
  _Float16* b16  = (_Float16*)take((size_t)BSc * DIc * 2);
  float*    xz   = (float*)   take((size_t)BSc * 2 * DIc * 4);
  float*    xc   = (float*)   take((size_t)BSc * DIc * 4);
  float*    dbc  = (float*)   take((size_t)BSc * DBCc * 4);
  float*    delta= (float*)   take((size_t)BSc * DIc * 4);
  float*    mout = (float*)   take((size_t)BSc * DMc * 4);

  auto blks = [](int n) { return (n + 255) / 256; };

  // ---- weight conversion (f32 -> f16, w_dt zero-padded to K=64) ----
  for (int l = 0; l < Lc; ++l) {
    _Float16* Lw = wf16 + (size_t)l * PLH;
    _Float16* Linproj = Lw;
    _Float16* Lxproj  = Linproj + (size_t)2 * DIc * DMc;
    _Float16* Ldtp    = Lxproj  + (size_t)DBCc * DIc;
    _Float16* Loutp   = Ldtp    + (size_t)DIc * DTRp;
    _Float16* Lffw1   = Loutp   + (size_t)DMc * DIc;
    _Float16* Lffw2   = Lffw1   + (size_t)FFH * DMc;
    f32_to_f16_k<<<blks(2*DIc*DMc), 256, 0, stream>>>(w_inproj + (size_t)l*2*DIc*DMc, Linproj, 2*DIc*DMc);
    f32_to_f16_k<<<blks(DBCc*DIc), 256, 0, stream>>>(w_xproj + (size_t)l*DBCc*DIc, Lxproj, DBCc*DIc);
    wdt_pad_k<<<blks(DIc*DTRp), 256, 0, stream>>>(w_dt + (size_t)l*DIc*DTRc, Ldtp);
    f32_to_f16_k<<<blks(DMc*DIc), 256, 0, stream>>>(w_outproj + (size_t)l*DMc*DIc, Loutp, DMc*DIc);
    f32_to_f16_k<<<blks(FFH*DMc), 256, 0, stream>>>(ff_w1 + (size_t)l*FFH*DMc, Lffw1, FFH*DMc);
    f32_to_f16_k<<<blks(DMc*FFH), 256, 0, stream>>>(ff_w2 + (size_t)l*DMc*FFH, Lffw2, DMc*FFH);
  }

  // ---- embedding + positional encoding ----
  embed_k<<<blks(BSc * DMc), 256, 0, stream>>>(x, w_in, b_in, pe, hbuf);

  for (int l = 0; l < Lc; ++l) {
    _Float16* Lw = wf16 + (size_t)l * PLH;
    _Float16* Linproj = Lw;
    _Float16* Lxproj  = Linproj + (size_t)2 * DIc * DMc;
    _Float16* Ldtp    = Lxproj  + (size_t)DBCc * DIc;
    _Float16* Loutp   = Ldtp    + (size_t)DIc * DTRp;
    _Float16* Lffw1   = Loutp   + (size_t)DMc * DIc;
    _Float16* Lffw2   = Lffw1   + (size_t)FFH * DMc;

    // LN1 -> f16 A
    layernorm_k<true><<<BSc, 256, 0, stream>>>(hbuf, ln1_w + l*DMc, ln1_b + l*DMc, a16, nullptr, DMc);
    // in-proj: (2048x768) x (3072x768)^T -> xz
    wmma_gemm_lds<0><<<dim3(2*DIc/128, BSc/64), 256, 0, stream>>>(a16, DMc, Linproj, DMc, xz, nullptr, nullptr, 2*DIc, DMc);
    // depthwise causal conv + SiLU on xm half
    conv_silu_k<<<blks(BSc * DIc), 256, 0, stream>>>(xz, conv_w + l*DIc*Kc, conv_b + l*DIc, xc);
    // xc -> f16
    f32_to_f16_k<<<blks(BSc * DIc), 256, 0, stream>>>(xc, b16, BSc * DIc);
    // x-proj: (2048x1536) x (80x1536)^T -> dbc  (N=80 edge kernel)
    wmma_gemm_edge<0><<<dim3(1, BSc/64), 256, 0, stream>>>(b16, DIc, Lxproj, DIc, dbc, nullptr, nullptr, DBCc, DIc);
    // dt slice -> padded f16 A (K=64)
    dt_pad_k<<<blks(BSc * DTRp), 256, 0, stream>>>(dbc, a16);
    // delta = softplus(dt @ w_dt^T + b_dt)
    wmma_gemm_lds<3><<<dim3(DIc/128, BSc/64), 256, 0, stream>>>(a16, DTRp, Ldtp, DTRp, delta, nullptr, b_dt + l*DIc, DIc, DTRp);
    // selective scan -> y (f16, gated, +u*D) into a16
    scan_k<<<blks(Bc * DIc), 256, 0, stream>>>(delta, dbc, xz, xc, A_log + l*DIc*DSc, Dp + l*DIc, a16);
    // out-proj: (2048x1536) x (768x1536)^T -> mout
    wmma_gemm_lds<0><<<dim3(DMc/128, BSc/64), 256, 0, stream>>>(a16, DIc, Loutp, DIc, mout, nullptr, nullptr, DMc, DIc);
    add_inplace_k<<<blks(BSc * DMc), 256, 0, stream>>>(hbuf, mout, BSc * DMc);

    // FFN
    layernorm_k<true><<<BSc, 256, 0, stream>>>(hbuf, ln2_w + l*DMc, ln2_b + l*DMc, a16, nullptr, DMc);
    // ff1: gelu(bias + act @ w1^T) written directly as f16
    wmma_gemm_lds<2><<<dim3(FFH/128, BSc/64), 256, 0, stream>>>(a16, DMc, Lffw1, DMc, nullptr, b16, ff_b1 + l*FFH, FFH, DMc);
    // ff2: bias + hidden @ w2^T -> mout
    wmma_gemm_lds<1><<<dim3(DMc/128, BSc/64), 256, 0, stream>>>(b16, FFH, Lffw2, FFH, mout, nullptr, ff_b2 + l*DMc, DMc, FFH);
    add_inplace_k<<<blks(BSc * DMc), 256, 0, stream>>>(hbuf, mout, BSc * DMc);
  }

  // final LN (f32) -> reuse xc, then mean over S and head projection
  layernorm_k<false><<<BSc, 256, 0, stream>>>(hbuf, lnf_w, lnf_b, nullptr, xc, DMc);
  pool_head_k<<<Bc, 256, 0, stream>>>(xc, w_head, b_head, (float*)d_out);
}
